// BBBRNN_56659208569479
// MI455X (gfx1250) — compile-verified, matrix operand order checked by reading, not run on hardware
//
#include <hip/hip_runtime.h>
#include <hip/hip_bf16.h>
#include <cstdint>

// ---------------------------------------------------------------------------
// BBB (Bayes-by-Backprop) LSTM forward for MI455X (gfx1250, wave32, WMMA+TDM).
//
//   T=512, B=64, I=1024, H=1024, G=4H=4096
//
// Pipeline:
//   1. sample weights  w = mean + eps * exp(0.5*logvar), cast to bf16
//   2. x_gates = X(bf16) * W_ih(bf16)^T + (b_ih + b_hh)      [32768 x 4096] f32
//      -> compute-bound GEMM, v_wmma_f32_16x16x32_bf16,
//         TDM (tensor_load_to_lds) double-buffered LDS tiles
//   3. for t in 0..511:  gates = x_gates[t] + h * W_hh^T  (WMMA GEMM, M=64)
//                        c,h   = lstm_cell(gates, c)      (pointwise)
// ---------------------------------------------------------------------------

#define T_SEQ 512
#define BATCH 64
#define ISZ   1024
#define HSZ   1024
#define GSZ   (4 * HSZ)   // 4096

typedef __attribute__((ext_vector_type(16))) __bf16 bf16x16;
typedef __attribute__((ext_vector_type(8)))  float  f32x8;
typedef __attribute__((ext_vector_type(4)))  unsigned int su32x4;
typedef __attribute__((ext_vector_type(8)))  unsigned int su32x8;

union FragU { bf16x16 v; uint4 q[2]; };

__device__ __forceinline__ __bf16 f2bf(float f) {
  __hip_bfloat16 h = __float2bfloat16(f);
  __bf16 r;
  __builtin_memcpy(&r, &h, sizeof(r));
  return r;
}

// Low 32 bits of a generic pointer into __shared__ are the LDS byte address
__device__ __forceinline__ unsigned int lds_byte_addr(const void* p) {
  return (unsigned int)(uintptr_t)p;
}

// ---------------------------------------------------------------------------
// TDM: issue TENSOR_LOAD_TO_LDS for a 2D bf16 tile [rows x 32] with row
// stride `row_stride_elems`, packed row-major into LDS (ld = 32 elements).
// D# group0/1 per CDNA5 ISA; groups 2/3
// zeroed (tile_dim2 = 0 -> unused). Tracked by TENSORcnt.
// Issued by one wave per workgroup; EXEC is ignored by TDM.
// ---------------------------------------------------------------------------
__device__ __forceinline__ void tdm_load_tile_bf16(unsigned int lds_addr,
                                                   const __bf16* gsrc,
                                                   unsigned int rows,
                                                   unsigned int row_stride_elems) {
  const unsigned long long ga = (unsigned long long)(uintptr_t)gsrc;
  su32x4 g0;
  g0[0] = 1u;                                             // count=1, user mode
  g0[1] = lds_addr;                                       // lds_addr[31:0]
  g0[2] = (unsigned int)ga;                               // global_addr[31:0]
  g0[3] = ((unsigned int)(ga >> 32) & 0x01FFFFFFu)        // global_addr[56:32]
          | (2u << 30);                                   // type=2 ("image")
  su32x8 g1;
  g1[0] = (1u << 16);              // wg_mask=0, data_size=1 (2 bytes)
  g1[1] = 0xFFFF0000u;             // abar_addr=0 | tensor_dim0[15:0]
  g1[2] = 0xFFFF7FFFu;             // tensor_dim0[31:16]=0x7FFF | tensor_dim1[15:0]
  g1[3] = 0x7FFFu | (32u << 16);   // tensor_dim1[31:16] | tile_dim0=32
  g1[4] = rows & 0xFFFFu;          // tile_dim1=rows, tile_dim2=0
  g1[5] = row_stride_elems;        // tensor_dim0_stride[31:0] (data_size units)
  g1[6] = 0u;                      // stride[47:32]=0, tensor_dim1_stride[15:0]=0
  g1[7] = 0u;
  const su32x4 gz = {0u, 0u, 0u, 0u};
  asm volatile("tensor_load_to_lds %0, %1, %2, %3"
               :
               : "s"(g0), "s"(g1), "s"(gz), "s"(gz)
               : "memory");
}

// 16-bit WMMA A/B operand fragment from an LDS tile with leading dim 32.
// Per ISA 7.12.2 (16-bit A-matrix 16x32): lane<16 -> row=lane, K in
// {0..7, 16..23}; lane>=16 -> row=lane-16, K in {8..15, 24..31}.
__device__ __forceinline__ bf16x16 lds_frag(const __bf16* tile, int row0, int lane) {
  const int r  = lane & 15;
  const int kb = (lane >> 4) << 3;  // 0 or 8
  const __bf16* p = tile + (row0 + r) * 32 + kb;
  FragU u;
  u.q[0] = *reinterpret_cast<const uint4*>(p);
  u.q[1] = *reinterpret_cast<const uint4*>(p + 16);
  return u.v;
}

__device__ __forceinline__ f32x8 wmma_bf16(bf16x16 a, bf16x16 b, f32x8 c) {
  // (neg_a, A, neg_b, B, c_mod, C, reuse_a, reuse_b)
  return __builtin_amdgcn_wmma_f32_16x16x32_bf16(false, a, false, b, (short)0, c,
                                                 false, false);
}

// 32-bit C/D 16x16 layout: VGPR r -> M=r (lanes 0-15) / M=8+r (lanes 16-31),
// N = lane % 16.
__device__ __forceinline__ void store_acc(float* C, int ldc, int m0, int n0,
                                          f32x8 acc, int lane) {
  const int col = n0 + (lane & 15);
  const int rb  = m0 + ((lane >> 4) << 3);
#pragma unroll
  for (int r = 0; r < 8; ++r) C[(size_t)(rb + r) * ldc + col] = acc[r];
}

// ---------------------------------------------------------------------------
// Elementwise prep kernels
// ---------------------------------------------------------------------------
__global__ void sample_matrix_kernel(const float* __restrict__ mean,
                                     const float* __restrict__ logvar,
                                     const float* __restrict__ eps,
                                     __bf16* __restrict__ out, int n) {
  int i = blockIdx.x * blockDim.x + threadIdx.x;
  if (i < n) out[i] = f2bf(mean[i] + eps[i] * __expf(0.5f * logvar[i]));
}

__global__ void sample_bias_kernel(const float* __restrict__ bm_ih,
                                   const float* __restrict__ blv_ih,
                                   const float* __restrict__ be_ih,
                                   const float* __restrict__ bm_hh,
                                   const float* __restrict__ blv_hh,
                                   const float* __restrict__ be_hh,
                                   float* __restrict__ bias, int n) {
  int i = blockIdx.x * blockDim.x + threadIdx.x;
  if (i < n) {
    float bi = bm_ih[i] + be_ih[i] * __expf(0.5f * blv_ih[i]);
    float bh = bm_hh[i] + be_hh[i] * __expf(0.5f * blv_hh[i]);
    bias[i] = bi + bh;
  }
}

__global__ void f32_to_bf16_kernel(const float* __restrict__ x,
                                   __bf16* __restrict__ y, int n) {
  int i = blockIdx.x * blockDim.x + threadIdx.x;
  if (i < n) y[i] = f2bf(x[i]);
}

__global__ void init_state_kernel(__bf16* __restrict__ h_b,
                                  float* __restrict__ c, int n) {
  int i = blockIdx.x * blockDim.x + threadIdx.x;
  if (i < n) { h_b[i] = f2bf(0.0f); c[i] = 0.0f; }
}

// ---------------------------------------------------------------------------
// GEMM 1: x_gates = X * W_ih^T + bias         (TDM double-buffered)
//   X: [32768,1024] bf16 (row-major, K contiguous)
//   W: [4096,1024]  bf16 (row n = column n of W^T, K contiguous)
//   C: [32768,4096] f32
// Block tile 128(M) x 128(N); 8 waves as 4(M) x 2(N); wave = 32x64 = 2x4 tiles.
// Wave 0 drives the TDM: per K-step it waits TENSORcnt for the current
// buffer, then (post-barrier) issues the next buffer's A/B tile descriptors,
// overlapping the DMA with this step's WMMAs.
// ---------------------------------------------------------------------------
__global__ __launch_bounds__(256)
void xproj_gemm_kernel(const __bf16* __restrict__ X,
                       const __bf16* __restrict__ W,
                       const float*  __restrict__ bias,
                       float* __restrict__ C) {
  __shared__ __bf16 As[2][128 * 32];
  __shared__ __bf16 Bs[2][128 * 32];

  const int tid  = threadIdx.x;
  const int lane = tid & 31;
  const int wv   = tid >> 5;
  const int wm   = wv >> 1;   // 0..3
  const int wn   = wv & 1;    // 0..1
  const int mBase = blockIdx.y * 128;
  const int nBase = blockIdx.x * 128;
  const int m0 = wm * 32;
  const int n0 = wn * 64;

  const f32x8 zero = {0.f, 0.f, 0.f, 0.f, 0.f, 0.f, 0.f, 0.f};
  f32x8 acc[2][4];
#pragma unroll
  for (int i = 0; i < 2; ++i)
#pragma unroll
    for (int j = 0; j < 4; ++j) acc[i][j] = zero;

  const __bf16* Arow = X + (size_t)mBase * ISZ;   // tile row base (K advances)
  const __bf16* Brow = W + (size_t)nBase * ISZ;

  if (wv == 0) {
    tdm_load_tile_bf16(lds_byte_addr(&As[0][0]), Arow, 128, ISZ);
    tdm_load_tile_bf16(lds_byte_addr(&Bs[0][0]), Brow, 128, ISZ);
  }

  int buf = 0;
  for (int kk = 0; kk < ISZ; kk += 32) {
    if (wv == 0) __builtin_amdgcn_s_wait_tensorcnt(0);  // current buf landed
    __syncthreads();                                    // ...visible to all
    if (wv == 0 && (kk + 32) < ISZ) {                   // prefetch next buf
      tdm_load_tile_bf16(lds_byte_addr(&As[buf ^ 1][0]), Arow + kk + 32, 128, ISZ);
      tdm_load_tile_bf16(lds_byte_addr(&Bs[buf ^ 1][0]), Brow + kk + 32, 128, ISZ);
    }

    bf16x16 af[2], bfm[4];
#pragma unroll
    for (int i = 0; i < 2; ++i) af[i] = lds_frag(&As[buf][0], m0 + i * 16, lane);
#pragma unroll
    for (int j = 0; j < 4; ++j) bfm[j] = lds_frag(&Bs[buf][0], n0 + j * 16, lane);
#pragma unroll
    for (int i = 0; i < 2; ++i)
#pragma unroll
      for (int j = 0; j < 4; ++j) acc[i][j] = wmma_bf16(af[i], bfm[j], acc[i][j]);

    buf ^= 1;
  }

  const int col16 = lane & 15;
#pragma unroll
  for (int i = 0; i < 2; ++i)
#pragma unroll
    for (int j = 0; j < 4; ++j) {
      const int gm = mBase + m0 + i * 16;
      const int gn = nBase + n0 + j * 16;
      const float bval = bias[gn + col16];
      f32x8 a = acc[i][j];
#pragma unroll
      for (int r = 0; r < 8; ++r) a[r] += bval;
      store_acc(C, GSZ, gm, gn, a, lane);
    }
}

// ---------------------------------------------------------------------------
// GEMM 2 (per timestep): gates = x_gates[t] + h * W_hh^T
//   Hb: [64,1024] bf16; W: [4096,1024] bf16; XG/Gout: [64,4096] f32
// Block tile 64(M) x 128(N); 8 waves as 2(M) x 4(N); wave = 32x32 = 2x2 tiles.
// Keeps the global_load -> ds_store staging path (+ global_prefetch_b8);
// W_hh (8 MB bf16) stays L2-resident across all 512 steps.
// ---------------------------------------------------------------------------
__global__ __launch_bounds__(256)
void hgemm_step_kernel(const __bf16* __restrict__ Hb,
                       const __bf16* __restrict__ W,
                       const float*  __restrict__ XG,
                       float* __restrict__ Gout) {
  __shared__ __bf16 As[64 * 32];
  __shared__ __bf16 Bs[128 * 32];

  const int tid  = threadIdx.x;
  const int lane = tid & 31;
  const int wv   = tid >> 5;
  const int wm   = wv & 1;    // 0..1
  const int wn   = wv >> 1;   // 0..3
  const int nBase = blockIdx.x * 128;
  const int m0 = wm * 32;
  const int n0 = wn * 32;

  const f32x8 zero = {0.f, 0.f, 0.f, 0.f, 0.f, 0.f, 0.f, 0.f};
  f32x8 acc[2][2];
#pragma unroll
  for (int i = 0; i < 2; ++i)
#pragma unroll
    for (int j = 0; j < 2; ++j) acc[i][j] = zero;

  const int ldRow = tid >> 1;        // B tile: 0..127; A tile (tid<128): 0..63
  const int ldOff = (tid & 1) << 4;

  for (int kk = 0; kk < HSZ; kk += 32) {
    const __bf16* gb = W + (size_t)(nBase + ldRow) * HSZ + kk + ldOff;
    uint4 b0 = *reinterpret_cast<const uint4*>(gb);
    uint4 b1 = *reinterpret_cast<const uint4*>(gb + 8);
    uint4 a0, a1;
    if (tid < 128) {
      const __bf16* ga = Hb + (size_t)ldRow * HSZ + kk + ldOff;
      a0 = *reinterpret_cast<const uint4*>(ga);
      a1 = *reinterpret_cast<const uint4*>(ga + 8);
    }
    if (kk + 32 < HSZ) __builtin_prefetch(gb + 32, 0, 1);  // global_prefetch_b8
    __syncthreads();
    *reinterpret_cast<uint4*>(&Bs[ldRow * 32 + ldOff])     = b0;
    *reinterpret_cast<uint4*>(&Bs[ldRow * 32 + ldOff + 8]) = b1;
    if (tid < 128) {
      *reinterpret_cast<uint4*>(&As[ldRow * 32 + ldOff])     = a0;
      *reinterpret_cast<uint4*>(&As[ldRow * 32 + ldOff + 8]) = a1;
    }
    __syncthreads();

    bf16x16 af[2], bfm[2];
#pragma unroll
    for (int i = 0; i < 2; ++i) af[i]  = lds_frag(As, m0 + i * 16, lane);
#pragma unroll
    for (int j = 0; j < 2; ++j) bfm[j] = lds_frag(Bs, n0 + j * 16, lane);
#pragma unroll
    for (int i = 0; i < 2; ++i)
#pragma unroll
      for (int j = 0; j < 2; ++j) acc[i][j] = wmma_bf16(af[i], bfm[j], acc[i][j]);
  }

  const int col16 = lane & 15;
#pragma unroll
  for (int i = 0; i < 2; ++i)
#pragma unroll
    for (int j = 0; j < 2; ++j) {
      const int gm = m0 + i * 16 + ((lane >> 4) << 3);
      const int gn = nBase + n0 + j * 16 + col16;
#pragma unroll
      for (int r = 0; r < 8; ++r) {
        const size_t idx = (size_t)(gm + r) * GSZ + gn;
        Gout[idx] = acc[i][j][r] + XG[idx];
      }
    }
}

// ---------------------------------------------------------------------------
// Pointwise LSTM cell: c' = sig(f)*c + sig(i)*tanh(g); h' = sig(o)*tanh(c')
// ---------------------------------------------------------------------------
__global__ void lstm_pointwise_kernel(const float* __restrict__ gates, // [64,4096]
                                      float* __restrict__ c,          // [64,1024]
                                      float* __restrict__ h_out,      // [64,1024] f32
                                      __bf16* __restrict__ h_b) {     // [64,1024] bf16
  int i = blockIdx.x * blockDim.x + threadIdx.x;   // 0..65535
  if (i >= BATCH * HSZ) return;
  const int b  = i >> 10;
  const int hx = i & (HSZ - 1);
  const float* g = gates + (size_t)b * GSZ;
  const float gi = g[hx];
  const float gf = g[HSZ + hx];
  const float gg = g[2 * HSZ + hx];
  const float go = g[3 * HSZ + hx];
  const float si = 1.f / (1.f + __expf(-gi));
  const float sf = 1.f / (1.f + __expf(-gf));
  const float so = 1.f / (1.f + __expf(-go));
  const float cn = sf * c[i] + si * tanhf(gg);
  const float hn = so * tanhf(cn);
  c[i]     = cn;
  h_out[i] = hn;
  h_b[i]   = f2bf(hn);
}

// ---------------------------------------------------------------------------
// Host-side orchestration
// ---------------------------------------------------------------------------
extern "C" void kernel_launch(void* const* d_in, const int* in_sizes, int n_in,
                              void* d_out, int out_size, void* d_ws, size_t ws_size,
                              hipStream_t stream) {
  (void)in_sizes; (void)n_in; (void)out_size; (void)ws_size;

  const float* x    = (const float*)d_in[0];
  const float* wim  = (const float*)d_in[1];
  const float* wilv = (const float*)d_in[2];
  const float* whm  = (const float*)d_in[3];
  const float* whlv = (const float*)d_in[4];
  const float* bim  = (const float*)d_in[5];
  const float* bilv = (const float*)d_in[6];
  const float* bhm  = (const float*)d_in[7];
  const float* bhlv = (const float*)d_in[8];
  const float* ewi  = (const float*)d_in[9];
  const float* ewh  = (const float*)d_in[10];
  const float* ebi  = (const float*)d_in[11];
  const float* ebh  = (const float*)d_in[12];
  float* out = (float*)d_out;

  char* ws = (char*)d_ws;
  size_t off = 0;
  auto alloc = [&](size_t bytes) -> void* {
    void* p = ws + off;
    off += (bytes + 255) & ~(size_t)255;
    return p;
  };

  __bf16* w_ih_b = (__bf16*)alloc((size_t)GSZ * ISZ * 2);          //   8 MB
  __bf16* w_hh_b = (__bf16*)alloc((size_t)GSZ * HSZ * 2);          //   8 MB
  float*  bias   = (float*) alloc((size_t)GSZ * 4);                //  16 KB
  __bf16* x_b    = (__bf16*)alloc((size_t)T_SEQ * BATCH * ISZ * 2);//  64 MB
  float*  xg     = (float*) alloc((size_t)T_SEQ * BATCH * GSZ * 4);// 512 MB
  float*  gates  = (float*) alloc((size_t)BATCH * GSZ * 4);        //   1 MB
  __bf16* h_b    = (__bf16*)alloc((size_t)BATCH * HSZ * 2);        // 128 KB
  float*  c_st   = (float*) alloc((size_t)BATCH * HSZ * 4);        // 256 KB

  // 1. reparameterized weight sampling -> bf16
  {
    const int n = GSZ * ISZ;
    sample_matrix_kernel<<<(n + 255) / 256, 256, 0, stream>>>(wim, wilv, ewi, w_ih_b, n);
  }
  {
    const int n = GSZ * HSZ;
    sample_matrix_kernel<<<(n + 255) / 256, 256, 0, stream>>>(whm, whlv, ewh, w_hh_b, n);
  }
  sample_bias_kernel<<<(GSZ + 255) / 256, 256, 0, stream>>>(bim, bilv, ebi,
                                                            bhm, bhlv, ebh, bias, GSZ);
  {
    const int n = T_SEQ * BATCH * ISZ;
    f32_to_bf16_kernel<<<(n + 255) / 256, 256, 0, stream>>>(x, x_b, n);
  }
  init_state_kernel<<<(BATCH * HSZ + 255) / 256, 256, 0, stream>>>(h_b, c_st,
                                                                   BATCH * HSZ);

  // 2. x_gates = X * W_ih^T + bias   (compute-bound, WMMA + TDM)
  {
    dim3 grid(GSZ / 128, (T_SEQ * BATCH) / 128);  // (32, 256)
    xproj_gemm_kernel<<<grid, 256, 0, stream>>>(x_b, w_ih_b, bias, xg);
  }

  // 3. sequential recurrence
  for (int t = 0; t < T_SEQ; ++t) {
    hgemm_step_kernel<<<GSZ / 128, 256, 0, stream>>>(
        h_b, w_hh_b, xg + (size_t)t * BATCH * GSZ, gates);
    lstm_pointwise_kernel<<<(BATCH * HSZ) / 256, 256, 0, stream>>>(
        gates, c_st, out + (size_t)t * BATCH * HSZ, h_b);
  }
}
